// Encoder_34282428957169
// MI455X (gfx1250) — compile-verified, matrix operand order checked by reading, not run on hardware
//
#include <hip/hip_runtime.h>
#include <hip/hip_fp16.h>
#include <stdint.h>

typedef __attribute__((ext_vector_type(16))) _Float16 v16h;
typedef __attribute__((ext_vector_type(8)))  _Float16 v8h;
typedef __attribute__((ext_vector_type(8)))  float    v8f;

#define V_   64
#define E_   128
#define H_   256
#define B_   64
#define T_   512
#define G4H  1024   // 4*H

#define MB_   16     // batch rows per block (one WMMA M-tile)
#define NTHR  512    // threads per block = 16 waves
#define GSTRIDE 1032 // padded row stride (halves) for gates LDS
#define HSTRIDE 264  // padded row stride (halves) for h LDS

__device__ __forceinline__ float sigm(float x) { return 1.0f / (1.0f + __expf(-x)); }

// ---------------------------------------------------------------------------
// Prep 1: embW[d][v][n] = sum_k emb[v,k] * W0[d][k,n] + b0[d][n]   (f32)
// Layer-0's input projection collapses to a V=64-entry lookup table.
// ---------------------------------------------------------------------------
__global__ void embw_kernel(const float* __restrict__ emb,
                            const float* __restrict__ W0f, const float* __restrict__ b0f,
                            const float* __restrict__ W0b, const float* __restrict__ b0b,
                            float* __restrict__ embW) {
  int tid = blockIdx.x * blockDim.x + threadIdx.x;      // 2*64*1024
  if (tid >= 2 * V_ * G4H) return;
  int n = tid % G4H;
  int v = (tid / G4H) % V_;
  int d = tid / (G4H * V_);
  const float* W = d ? W0b : W0f;
  const float* b = d ? b0b : b0f;
  float acc = b[n];
  #pragma unroll 4
  for (int k = 0; k < E_; ++k) acc += emb[v * E_ + k] * W[k * G4H + n];
  embW[tid] = acc;
}

// Pack a K x 1024 f32 matrix into WMMA-B fragment-major f16 layout:
//   dst[((kt*64 + nt)*32 + lane)*16 + e]  holds  src[kbase + klocal, nt*16 + (lane&15)]
//   klocal = (e&7) + 8*(lane>>4) + 16*(e>>3)   (per ISA 16-bit A/B layout)
__device__ __forceinline__ void pack_frag_entry(const float* __restrict__ src,
                                                _Float16* __restrict__ dst,
                                                int entry, int kbase) {
  int lane = entry & 31;
  int nt   = (entry >> 5) & 63;
  int half = lane >> 4;
  int n    = nt * 16 + (lane & 15);
  #pragma unroll
  for (int e = 0; e < 16; ++e) {
    int klocal = (e & 7) + 8 * half + 16 * (e >> 3);
    int k = kbase + klocal;
    dst[(size_t)entry * 16 + e] = (_Float16)src[(size_t)k * G4H + n];
  }
}

// Prep 2: pack U0 (K=256 -> 8 ktiles) for both directions.
__global__ void pack_u0_kernel(const float* __restrict__ U0f, const float* __restrict__ U0b,
                               _Float16* __restrict__ U0pk) {
  int tid = blockIdx.x * blockDim.x + threadIdx.x;      // 2*8*64*32 = 32768
  if (tid >= 2 * 8 * 2048) return;
  int kt = (tid >> 11) & 7;
  int d  = tid >> 14;
  pack_frag_entry(d ? U0b : U0f, U0pk, tid, kt * 32);
}

// Prep 3: pack combined [W1 (K=512 -> ktiles 0..15) ; U1 (K=256 -> ktiles 16..23)].
__global__ void pack_b1_kernel(const float* __restrict__ W1f, const float* __restrict__ U1f,
                               const float* __restrict__ W1b, const float* __restrict__ U1b,
                               _Float16* __restrict__ B1pk) {
  int tid = blockIdx.x * blockDim.x + threadIdx.x;      // 2*24*64*32 = 98304
  if (tid >= 2 * 24 * 2048) return;
  int kt = (tid >> 11) % 24;
  int d  = tid / (24 * 2048);
  const float* src;
  int kbase;
  if (kt < 16) { src = d ? W1b : W1f; kbase = kt * 32; }
  else         { src = d ? U1b : U1f; kbase = (kt - 16) * 32; }
  pack_frag_entry(src, B1pk, tid, kbase);
}

__device__ __forceinline__ v16h load_afrag(const _Float16* p) {
  v8h lo = *(const v8h*)(p);
  v8h hi = *(const v8h*)(p + 16);
  return __builtin_shufflevector(lo, hi, 0,1,2,3,4,5,6,7,8,9,10,11,12,13,14,15);
}

// ---------------------------------------------------------------------------
// Persistent LSTM layer slice: one block = (direction, batch-group of 16 rows).
// Recurrence is independent per batch row -> batch-split multiplies WGP
// utilization on the sequential critical path with zero communication.
// 16 waves; each wave owns 64 gate-columns = 4 WMMA N-tiles (one gate type).
// B-fragments stream from the L2-resident packed weights every step; an
// opaque per-iteration SGPR offset stops LLVM hoisting them (which spilled)
// while preserving global address-space inference (global_load, not flat).
// ---------------------------------------------------------------------------
template <int LAYER, int KTILES, int KSPLIT>
__global__ __launch_bounds__(NTHR)
void lstm_kernel(const int*       __restrict__ x,       // (B,T)
                 const _Float16*  __restrict__ Bpk,     // [2][KTILES][64][32][16]
                 const float*     __restrict__ embW,    // layer0: [2][64][1024]
                 const float*     __restrict__ b1f,
                 const float*     __restrict__ b1b,     // layer1 biases
                 const _Float16*  __restrict__ y0,      // layer1 input (B,T,512) f16
                 _Float16*        __restrict__ y0out,   // layer0 output (B,T,512) f16
                 float*           __restrict__ out) {   // d_out
  extern __shared__ char smem[];
  float*    cbuf  = (float*)smem;                          // 16*256 f32
  float*    hf32  = cbuf + MB_ * H_;                       // 16*256 f32
  _Float16* gates = (_Float16*)(hf32 + MB_ * H_);          // 16*GSTRIDE f16
  _Float16* hbuf  = gates + MB_ * GSTRIDE;                 // 16*HSTRIDE f16
  int*      xcol  = (int*)(hbuf + MB_ * HSTRIDE);          // 16 ints

  const int dir  = blockIdx.x;       // 0 fwd, 1 bwd
  const int bg   = blockIdx.y;       // batch group: rows bg*16 .. bg*16+15
  const int tid  = threadIdx.x;
  const int lane = tid & 31;
  const int wave = tid >> 5;         // 0..15
  const int half = lane >> 4;
  const int l16  = lane & 15;
  const int gate = wave >> 2;        // 64 cols/wave -> gate = wave/4 (0:i 1:f 2:g 3:o)
  const int nbase = wave * 64;

  // zero-init recurrent state
  for (int i = tid; i < MB_ * H_; i += NTHR) { cbuf[i] = 0.0f; hf32[i] = 0.0f; }
  for (int i = tid; i < MB_ * HSTRIDE; i += NTHR) hbuf[i] = (_Float16)0.0f;

  float bias[4] = {0.f, 0.f, 0.f, 0.f};
  if (LAYER == 1) {
    const float* bb = dir ? b1b : b1f;
    #pragma unroll
    for (int nt = 0; nt < 4; ++nt) bias[nt] = bb[nbase + nt * 16 + l16];
  }

  // lane-resolved base into packed B: + (ktile,ntile) offsets added per use
  const _Float16* Bp = Bpk + (size_t)dir * KTILES * 2048 * 16
                           + ((size_t)wave * 4 * 32 + lane) * 16;
  const float*    eW = (LAYER == 0) ? (embW + (size_t)dir * V_ * G4H) : nullptr;

  for (int t = 0; t < T_; ++t) {
    const int tsrc = dir ? (T_ - 1 - t) : t;

    // Opaque (always-zero) scalar offset generated fresh each iteration:
    // defeats LICM on the t-invariant B-fragment loads without losing the
    // global address space of Bp (keeps global_load_b128 / SADDR form).
    uint64_t zoff = 0;
    asm volatile("" : "+s"(zoff));
    const _Float16* Bt = Bp + zoff;

    __syncthreads();                       // prev-step h/c/gates settled
    if (tid < MB_) xcol[tid] = x[(bg * MB_ + tid) * T_ + tsrc];
    __syncthreads();

    // ---- accumulator init: gathered embW (layer0) or bias splat (layer1)
    v8f acc[4];
    if (LAYER == 0) {
      #pragma unroll
      for (int v = 0; v < 8; ++v) {
        int tok = xcol[v + 8 * half];
        #pragma unroll
        for (int nt = 0; nt < 4; ++nt)
          acc[nt][v] = eW[(size_t)tok * G4H + nbase + nt * 16 + l16];
      }
    } else {
      #pragma unroll
      for (int nt = 0; nt < 4; ++nt)
        #pragma unroll
        for (int v = 0; v < 8; ++v) acc[nt][v] = bias[nt];
    }

    // ---- K reduction part 1: A from global y0 (layer-1 input projection)
    #pragma unroll
    for (int kt = 0; kt < KSPLIT; ++kt) {
      v16h a = load_afrag(y0 + ((size_t)(bg * MB_ + l16) * T_ + tsrc) * 512
                             + kt * 32 + 8 * half);
      const _Float16* bptr = Bt + (size_t)kt * 64 * 32 * 16;
      v16h bf[4];
      #pragma unroll
      for (int nt = 0; nt < 4; ++nt) bf[nt] = *(const v16h*)(bptr + (size_t)nt * 512);
      #pragma unroll
      for (int nt = 0; nt < 4; ++nt)
        acc[nt] = __builtin_amdgcn_wmma_f32_16x16x32_f16(false, a, false, bf[nt],
                                                         (short)0, acc[nt], false, false);
    }

    // ---- K reduction part 2: A from recurrent h in LDS
    #pragma unroll
    for (int kt = KSPLIT; kt < KTILES; ++kt) {
      v16h a = load_afrag(hbuf + l16 * HSTRIDE + (kt - KSPLIT) * 32 + 8 * half);
      const _Float16* bptr = Bt + (size_t)kt * 64 * 32 * 16;
      v16h bf[4];
      #pragma unroll
      for (int nt = 0; nt < 4; ++nt) bf[nt] = *(const v16h*)(bptr + (size_t)nt * 512);
      #pragma unroll
      for (int nt = 0; nt < 4; ++nt)
        acc[nt] = __builtin_amdgcn_wmma_f32_16x16x32_f16(false, a, false, bf[nt],
                                                         (short)0, acc[nt], false, false);
    }

    // ---- activation + scatter to gates LDS (C layout: lane->N, v+8*half->M)
    #pragma unroll
    for (int nt = 0; nt < 4; ++nt) {
      int n = nbase + nt * 16 + l16;
      #pragma unroll
      for (int v = 0; v < 8; ++v) {
        int b = v + 8 * half;
        float z = acc[nt][v];
        float a = (gate == 2) ? tanhf(z) : sigm(z);
        gates[b * GSTRIDE + n] = (_Float16)a;
      }
    }
    __syncthreads();

    // ---- elementwise LSTM cell update (8 elements / thread, 16*256 total)
    #pragma unroll
    for (int e = 0; e < 8; ++e) {
      int idx = e * NTHR + tid;
      int b = idx >> 8, j = idx & 255;
      float gi = (float)gates[b * GSTRIDE + j];
      float gf = (float)gates[b * GSTRIDE + 256 + j];
      float gg = (float)gates[b * GSTRIDE + 512 + j];
      float go = (float)gates[b * GSTRIDE + 768 + j];
      float c  = cbuf[idx];
      float cn = gf * c + gi * gg;
      float hn = go * tanhf(cn);
      bool  m  = (xcol[b] != 0);
      float c2 = m ? cn : c;
      float h2 = m ? hn : hf32[idx];
      cbuf[idx] = c2;
      hf32[idx] = h2;
      hbuf[b * HSTRIDE + j] = (_Float16)h2;
      size_t oidx = ((size_t)(bg * MB_ + b) * T_ + tsrc) * 512 + dir * 256 + j;
      if (LAYER == 0) y0out[oidx] = (_Float16)h2;
      else            out[oidx]   = h2;
    }
  }

  __syncthreads();
  // ---- final state outputs: sh/sc (B, 2H) at the tail of d_out
  const size_t OFF = (size_t)B_ * T_ * 512;                // end of y1
  size_t shb = OFF + (size_t)(LAYER ? 2 : 0) * 32768;      // sh0 or sh1
  size_t scb = shb + 32768;                                // sc0 or sc1
  #pragma unroll
  for (int e = 0; e < 8; ++e) {
    int idx = e * NTHR + tid;
    int b = idx >> 8, j = idx & 255;
    out[shb + (size_t)(bg * MB_ + b) * 512 + dir * 256 + j] = hf32[idx];
    out[scb + (size_t)(bg * MB_ + b) * 512 + dir * 256 + j] = cbuf[idx];
  }
}

// ---------------------------------------------------------------------------
extern "C" void kernel_launch(void* const* d_in, const int* in_sizes, int n_in,
                              void* d_out, int out_size, void* d_ws, size_t ws_size,
                              hipStream_t stream) {
  const int*   x   = (const int*)d_in[0];
  const float* emb = (const float*)d_in[1];
  const float* W0f = (const float*)d_in[2];
  const float* U0f = (const float*)d_in[3];
  const float* b0f = (const float*)d_in[4];
  const float* W0b = (const float*)d_in[5];
  const float* U0b = (const float*)d_in[6];
  const float* b0b = (const float*)d_in[7];
  const float* W1f = (const float*)d_in[8];
  const float* U1f = (const float*)d_in[9];
  const float* b1f = (const float*)d_in[10];
  const float* W1b = (const float*)d_in[11];
  const float* U1b = (const float*)d_in[12];
  const float* b1b = (const float*)d_in[13];
  float* out = (float*)d_out;

  // workspace carve-up
  char* ws = (char*)d_ws;
  float*    embW = (float*)ws;                                   //   512 KB
  _Float16* U0pk = (_Float16*)(ws + 524288);                     //  1 MB
  _Float16* B1pk = (_Float16*)(ws + 524288 + 1048576);           //  3 MB
  _Float16* y0   = (_Float16*)(ws + 524288 + 1048576 + 3145728); // 32 MB

  embw_kernel<<<512, 256, 0, stream>>>(emb, W0f, b0f, W0b, b0b, embW);
  pack_u0_kernel<<<128, 256, 0, stream>>>(U0f, U0b, U0pk);
  pack_b1_kernel<<<384, 256, 0, stream>>>(W1f, U1f, W1b, U1b, B1pk);

  const size_t smem = (size_t)MB_ * H_ * 4 * 2         // cbuf + hf32
                    + (size_t)MB_ * GSTRIDE * 2        // gates
                    + (size_t)MB_ * HSTRIDE * 2        // hbuf
                    + 64;                              // xcol

  (void)hipFuncSetAttribute(reinterpret_cast<const void*>(lstm_kernel<0, 8, 0>),
                            hipFuncAttributeMaxDynamicSharedMemorySize, (int)smem);
  (void)hipFuncSetAttribute(reinterpret_cast<const void*>(lstm_kernel<1, 24, 16>),
                            hipFuncAttributeMaxDynamicSharedMemorySize, (int)smem);

  dim3 grid(2, B_ / MB_);   // (direction, batch group) = 8 independent WGPs

  // layer 0: A entirely from recurrent h (KSPLIT=0)
  lstm_kernel<0, 8, 0><<<grid, NTHR, smem, stream>>>(x, U0pk, embW, nullptr, nullptr,
                                                     nullptr, y0, out);
  // layer 1: fused y0@W1 (ktiles 0..15 from global y0) + h@U1 (16..23 from LDS)
  lstm_kernel<1, 24, 16><<<grid, NTHR, smem, stream>>>(x, B1pk, nullptr, b1f, b1b,
                                                       y0, nullptr, out);
}
